// MultiHeadAttentionLayer_85504208928874
// MI455X (gfx1250) — matrix operand order repaired and checked by reading, not compiled, not run on hardware
//
#include <hip/hip_runtime.h>
#include <hip/hip_bf16.h>

// GAT-style multi-head attention layer for MI455X (gfx1250, wave32).
// Phase 0: convert h -> f16 and W{q,k,v} -> transposed f16 (tiny, one pass).
// Phase 1: fused Q/K/V projections via v_wmma_f32_16x16x32_f16 (f32 accum,
//          bias in f32). One wave owns a 16-row strip of one matrix and
//          reuses its A fragments across all 8 column tiles (32 WMMAs/wave).
//          Outputs stored f16: 38.4MB of Q/K/V << 192MB L2 for the gathers.
// Phase 2: per-edge scores + clipped exp + f32 atomic scatter (the real
//          bottleneck; ~1GB of L2-resident traffic + atomics).
// Phase 3: normalize.

typedef __attribute__((ext_vector_type(16))) _Float16 v16h;
typedef __attribute__((ext_vector_type(8)))  _Float16 v8h;
typedef __attribute__((ext_vector_type(4)))  _Float16 v4h;
typedef __attribute__((ext_vector_type(4)))  float    v4f;
typedef __attribute__((ext_vector_type(8)))  float    v8f;

#define IN_DIM 128
#define HD     128          // H * D
#define NHEAD  8
#define INV_SCALE 0.25f     // 1/sqrt(16)

// ---------------------------------------------------------------------------
// Phase 0a: h (fp32) -> h16 (f16), vectorized 4 elements/thread.
// ---------------------------------------------------------------------------
__global__ void prep_h16(const float* __restrict__ h, _Float16* __restrict__ h16,
                         int nvec /* = n*128/4 */)
{
    const int i = blockIdx.x * blockDim.x + threadIdx.x;
    if (i >= nvec) return;
    const v4f x = ((const v4f*)h)[i];
    v4h o;
    o[0] = (_Float16)x[0]; o[1] = (_Float16)x[1];
    o[2] = (_Float16)x[2]; o[3] = (_Float16)x[3];
    ((v4h*)h16)[i] = o;
}

// ---------------------------------------------------------------------------
// Phase 0b: W{q,k,v} (fp32 [K=128][N=128]) -> Wt16[which][col][k] (f16),
// i.e. transposed, so each 8-K half of a B fragment (one column) is one
// contiguous 16-byte load. 3*128*128 = 49152 elements total.
// ---------------------------------------------------------------------------
__global__ void prep_wT(const float* __restrict__ Wq,
                        const float* __restrict__ Wk,
                        const float* __restrict__ Wv,
                        _Float16* __restrict__ Wt16)
{
    const int t = blockIdx.x * blockDim.x + threadIdx.x;
    if (t >= 3 * 128 * 128) return;
    const int which = t >> 14;           // /16384
    const int col   = (t >> 7) & 127;
    const int k     = t & 127;
    const float* W = (which == 0) ? Wq : (which == 1) ? Wk : Wv;
    Wt16[t] = (_Float16)W[(size_t)k * HD + col];
}

// ---------------------------------------------------------------------------
// Phase 1: one wave = one 16-row strip x one matrix (Q/K/V). Loads the 4 A
// fragments once (32 VGPRs), then loops 8 column tiles x 4 k-steps of
// v_wmma_f32_16x16x32_f16.
// Fragment layouts per CDNA5 ISA 7.12.2 (wave32):
//   A (16x32 f16): lane l -> M = l&15, half = l>>4;
//       elem j -> K = (j<8 ? half*8 + j : 16 + half*8 + (j-8))
//   B (32x16 f16): lane l -> N = l&15; elem j -> K same as A (half = l>>4)
//   C/D (16x16 f32): vgpr r -> M = (l>>4)*8 + r, N = l&15
// ---------------------------------------------------------------------------
__global__ void qkv_gemm_wmma(const _Float16* __restrict__ h16,
                              const _Float16* __restrict__ Wt16,
                              const float* __restrict__ bq,
                              const float* __restrict__ bk,
                              const float* __restrict__ bv,
                              _Float16* __restrict__ Qh,
                              _Float16* __restrict__ Kh,
                              _Float16* __restrict__ Vh,
                              int n)
{
    const int gwave = (blockIdx.x * blockDim.x + threadIdx.x) >> 5;
    const int lane  = threadIdx.x & 31;

    const int mtiles = (n + 15) >> 4;
    if (gwave >= mtiles * 3) return;       // whole wave exits (EXEC stays full)

    const int mt    = gwave / 3;
    const int which = gwave % 3;

    const float* __restrict__ bb = (which == 0) ? bq : (which == 1) ? bk : bv;
    _Float16*    __restrict__ Ot = (which == 0) ? Qh : (which == 1) ? Kh : Vh;
    const _Float16* __restrict__ Wt = Wt16 + (size_t)which * (128 * 128);

    const int half = lane >> 4;
    const int l15  = lane & 15;
    const int arow = mt * 16 + l15;
    const int arwc = (arow < n) ? arow : (n - 1);

    // wave-uniform: is this a full 16-row tile? (always true for n%16==0)
    const bool full = (mt * 16 + 16 <= n);

    // ---- A fragments for all 4 k-blocks, held in registers ----
    v16h a[4];
    #pragma unroll
    for (int kb = 0; kb < 4; ++kb) {
        const _Float16* ap = h16 + (size_t)arwc * IN_DIM + kb * 32 + half * 8;
        const v8h lo = *(const v8h*)ap;          // K = kb*32 + half*8 + 0..7
        const v8h hi = *(const v8h*)(ap + 16);   // K = kb*32 + 16 + half*8 + 0..7
        a[kb] = __builtin_shufflevector(lo, hi, 0, 1, 2, 3, 4, 5, 6, 7,
                                                8, 9, 10, 11, 12, 13, 14, 15);
    }

    for (int nt = 0; nt < 8; ++nt) {
        const int ncol = nt * 16 + l15;
        v8f acc = {};
        #pragma unroll
        for (int kb = 0; kb < 4; ++kb) {
            // B fragment: K = kb*32 + half*8 + 0..7, then kb*32 + 16 + half*8 + 0..7 (matches A)
            const _Float16* bp = Wt + (size_t)ncol * 128 + kb * 32 + half * 8;
            const v8h blo = *(const v8h*)bp;
            const v8h bhi = *(const v8h*)(bp + 16);
            const v16h b = __builtin_shufflevector(blo, bhi, 0, 1, 2, 3, 4, 5, 6, 7,
                                                   8, 9, 10, 11, 12, 13, 14, 15);
            acc = __builtin_amdgcn_wmma_f32_16x16x32_f16(
                      /*neg_a=*/false, a[kb], /*neg_b=*/false, b,
                      /*c_mod=*/(short)0, acc, /*reuse_a=*/false, /*reuse_b=*/false);
        }
        const float bias = bb[ncol];
        _Float16* op = Ot + (size_t)(mt * 16 + half * 8) * HD + ncol;
        if (full) {
            // uniform fast path: no per-store EXEC juggling
            #pragma unroll
            for (int r = 0; r < 8; ++r)
                op[(size_t)r * HD] = (_Float16)(acc[r] + bias);
        } else {
            #pragma unroll
            for (int r = 0; r < 8; ++r) {
                const int m = mt * 16 + half * 8 + r;
                if (m < n)
                    Ot[(size_t)m * HD + ncol] = (_Float16)(acc[r] + bias);
            }
        }
    }
}

// ---------------------------------------------------------------------------
// Phase 2: 128 threads per edge (one per head x dim). 16-lane shuffle
// reduction inside the wave32 gives the per-head dot product; then clipped
// exp and f32 atomic scatter-add of V*score and score.
// ---------------------------------------------------------------------------
__global__ void edge_scatter(const _Float16* __restrict__ Qh,
                             const _Float16* __restrict__ Kh,
                             const _Float16* __restrict__ Vh,
                             const float* __restrict__ dis,
                             const float* __restrict__ att_w,
                             const float* __restrict__ att_b,
                             const int* __restrict__ src,
                             const int* __restrict__ dst,
                             float* __restrict__ wV,
                             float* __restrict__ z,
                             int nedges)
{
    const long long gid = (long long)blockIdx.x * blockDim.x + threadIdx.x;
    const int e = (int)(gid >> 7);
    if (e >= nedges) return;
    const int lane = (int)(gid & 127);     // head*16 + d
    const int hh   = lane >> 4;

    const int s = src[e];
    const int t = dst[e];

    const float q = (float)Qh[(size_t)t * HD + lane];
    const float k = (float)Kh[(size_t)s * HD + lane];
    float p = q * k;
    p += __shfl_xor(p, 1);
    p += __shfl_xor(p, 2);
    p += __shfl_xor(p, 4);
    p += __shfl_xor(p, 8);

    const float bias = dis[t] * att_w[hh] + att_b[hh];
    float sc = (p + bias) * INV_SCALE;
    sc = fminf(fmaxf(sc, -5.0f), 5.0f);
    const float w = __expf(sc);

    const float v = (float)Vh[(size_t)s * HD + lane];
    atomicAdd(&wV[(size_t)t * HD + lane], v * w);
    if ((lane & 15) == 0)
        atomicAdd(&z[(size_t)t * NHEAD + hh], w);
}

// ---------------------------------------------------------------------------
// Phase 3: out[n,h,d] = wV[n,h,d] / z[n,h]
// ---------------------------------------------------------------------------
__global__ void normalize_out(const float* __restrict__ wV,
                              const float* __restrict__ z,
                              float* __restrict__ out,
                              int n)
{
    const int i = blockIdx.x * blockDim.x + threadIdx.x;
    if (i >= n * HD) return;
    const int node = i >> 7;
    const int hh   = (i >> 4) & 7;
    out[i] = wV[i] / z[(size_t)node * NHEAD + hh];
}

extern "C" void kernel_launch(void* const* d_in, const int* in_sizes, int n_in,
                              void* d_out, int out_size, void* d_ws, size_t ws_size,
                              hipStream_t stream)
{
    const float* h   = (const float*)d_in[0];
    const float* dis = (const float*)d_in[1];
    const float* Wq  = (const float*)d_in[2];
    const float* bq  = (const float*)d_in[3];
    const float* Wk  = (const float*)d_in[4];
    const float* bk  = (const float*)d_in[5];
    const float* Wv  = (const float*)d_in[6];
    const float* bv  = (const float*)d_in[7];
    const float* aw  = (const float*)d_in[8];
    const float* ab  = (const float*)d_in[9];
    const int*   src = (const int*)d_in[10];
    const int*   dst = (const int*)d_in[11];
    float* out = (float*)d_out;

    const int n = in_sizes[0] / IN_DIM;     // 50000
    const int E = in_sizes[10];             // 800000

    // Workspace layout (~78.5 MB):
    //   Qh,Kh,Vh : n*128 f16 each   (38.4 MB, L2-resident for the gathers)
    //   wV       : n*128 f32        (25.6 MB)
    //   z        : n*8   f32        ( 1.6 MB)
    //   Wt16     : 3*128*128 f16    (96 KB, transposed weights)
    //   h16      : n*128 f16        (12.8 MB)
    _Float16* Qh   = (_Float16*)d_ws;
    _Float16* Kh   = Qh + (size_t)n * HD;
    _Float16* Vh   = Kh + (size_t)n * HD;
    float*    wV   = (float*)(Vh + (size_t)n * HD);
    float*    z    = wV + (size_t)n * HD;
    _Float16* Wt16 = (_Float16*)(z + (size_t)n * NHEAD);
    _Float16* h16  = Wt16 + (size_t)3 * 128 * 128;

    // zero accumulators every call (deterministic; graph-capture safe)
    hipMemsetAsync(wV, 0, ((size_t)n * HD + (size_t)n * NHEAD) * sizeof(float), stream);

    // Phase 0: conversions
    const int nvec = (n * IN_DIM) / 4;
    prep_h16<<<(nvec + 255) / 256, 256, 0, stream>>>(h, h16, nvec);
    prep_wT<<<(3 * 128 * 128 + 255) / 256, 256, 0, stream>>>(Wq, Wk, Wv, Wt16);

    // Phase 1: 3125 M-tiles x 3 matrices = 9375 waves, 8 waves/block
    const int mtiles = (n + 15) >> 4;
    const int waves  = mtiles * 3;
    const int gblk   = (waves + 7) >> 3;
    qkv_gemm_wmma<<<gblk, 256, 0, stream>>>(h16, Wt16, bq, bk, bv,
                                            Qh, Kh, Vh, n);

    // Phase 2: 128 threads per edge
    const long long ethreads = (long long)E * 128;
    const int eblk = (int)((ethreads + 255) / 256);
    edge_scatter<<<eblk, 256, 0, stream>>>(Qh, Kh, Vh, dis, aw, ab,
                                           src, dst, wV, z, E);

    // Phase 3
    const int fthreads = n * HD;
    normalize_out<<<(fthreads + 255) / 256, 256, 0, stream>>>(wV, z, out, n);
}
